// Model_11613591568823
// MI455X (gfx1250) — compile-verified, matrix operand order checked by reading, not run on hardware
//
#include <hip/hip_runtime.h>

// ---------------- problem constants ----------------
#define N_OPS   1024
#define OP_DIM  512
#define EMBED   1024
#define HALF_D  512
#define E_DSP   32768
#define N_SLOTS 256
#define SLOT_DIM 64
#define EDGE_DIM 16
#define E_RES   8192
#define MAXD    128

typedef _Float16 h16 __attribute__((ext_vector_type(16)));
typedef float    f8  __attribute__((ext_vector_type(8)));

// ---------------- fragment helpers ----------------
template <int BASE>
__device__ __forceinline__ void load8(h16& f, const float* __restrict__ p) {
    const float4 u = ((const float4*)p)[0];
    const float4 v = ((const float4*)p)[1];
    f[BASE + 0] = (_Float16)u.x; f[BASE + 1] = (_Float16)u.y;
    f[BASE + 2] = (_Float16)u.z; f[BASE + 3] = (_Float16)u.w;
    f[BASE + 4] = (_Float16)v.x; f[BASE + 5] = (_Float16)v.y;
    f[BASE + 6] = (_Float16)v.z; f[BASE + 7] = (_Float16)v.w;
}
template <int BASE>
__device__ __forceinline__ void zero8(h16& f) {
#pragma unroll
    for (int j = 0; j < 8; ++j) f[BASE + j] = (_Float16)0.f;
}

#define WMMA4(acc, a0, a1, b0, b1)                                                                        \
    acc[0][0] = __builtin_amdgcn_wmma_f32_16x16x32_f16(false, a0, false, b0, (short)0, acc[0][0], false, false); \
    acc[0][1] = __builtin_amdgcn_wmma_f32_16x16x32_f16(false, a0, false, b1, (short)0, acc[0][1], false, false); \
    acc[1][0] = __builtin_amdgcn_wmma_f32_16x16x32_f16(false, a1, false, b0, (short)0, acc[1][0], false, false); \
    acc[1][1] = __builtin_amdgcn_wmma_f32_16x16x32_f16(false, a1, false, b1, (short)0, acc[1][1], false, false)

// Full 32-wide K-chunk: pointers already include row base + half*8 lane offset.
__device__ __forceinline__ void wmma_chunk(f8 (&acc)[2][2],
        const float* __restrict__ pa0, const float* __restrict__ pa1,
        const float* __restrict__ pw0, const float* __restrict__ pw1)
{
    h16 a0, a1, b0, b1;
    load8<0>(a0, pa0); load8<8>(a0, pa0 + 16);
    load8<0>(a1, pa1); load8<8>(a1, pa1 + 16);
    load8<0>(b0, pw0); load8<8>(b0, pw0 + 16);
    load8<0>(b1, pw1); load8<8>(b1, pw1 + 16);
    WMMA4(acc, a0, a1, b0, b1);
}
// Tail half-chunk (remaining K2 % 32 == 16): upper fragment half is zero.
__device__ __forceinline__ void wmma_chunk_half(f8 (&acc)[2][2],
        const float* __restrict__ pa0, const float* __restrict__ pa1,
        const float* __restrict__ pw0, const float* __restrict__ pw1)
{
    h16 a0, a1, b0, b1;
    load8<0>(a0, pa0); zero8<8>(a0);
    load8<0>(a1, pa1); zero8<8>(a1);
    load8<0>(b0, pw0); zero8<8>(b0);
    load8<0>(b1, pw1); zero8<8>(b1);
    WMMA4(acc, a0, a1, b0, b1);
}

// =====================================================================
// WMMA GEMM:  OUT[M,N] = act( Arows[M,K] @ W[N,K]^T + bias )
//   A row r = concat( A1[ gather? gather[r] : r , :K1 ],  A2[ r , :K2 ] )
//   epilogue: store to OUT, or atomic segment-sum into segOut[seg[r]*N + n]
// Requirements met by all call sites: M%32==0, N%32==0, K1%32==0, K2%16==0.
// 2x2 tiles (32x32 output) per wave; branch-free phase loops.
// =====================================================================
__global__ __launch_bounds__(256) void wmma_gemm(
    const float* __restrict__ A1, int K1,
    const float* __restrict__ A2, int K2,
    const int*   __restrict__ gather,
    const float* __restrict__ W,
    const float* __restrict__ bias,
    int M, int N,
    float* __restrict__ OUT, int ldOut, int outColOff,
    const int* __restrict__ seg, float* __restrict__ segOut,
    int relu)
{
    const int K = K1 + K2;
    const int tilesN = N >> 5;
    const int tilesM = M >> 5;
    const int wave = threadIdx.x >> 5;
    const int lane = threadIdx.x & 31;
    const int tile = blockIdx.x * 8 + wave;
    if (tile >= tilesM * tilesN) return;       // wave-uniform exit
    const int tM = tile / tilesN;
    const int tN = tile % tilesN;

    const int half = lane >> 4;
    const int ml   = lane & 15;
    const int ho   = half * 8;                 // lane K-offset (address only!)

    const int gm0 = tM * 32 + ml;
    const int gm1 = gm0 + 16;
    const long rA0 = (gather ? (long)gather[gm0] : (long)gm0) * K1;
    const long rA1 = (gather ? (long)gather[gm1] : (long)gm1) * K1;
    const long rB0 = (long)gm0 * K2;
    const long rB1 = (long)gm1 * K2;

    const int gn0 = tN * 32 + ml;
    const int gn1 = gn0 + 16;
    const float* __restrict__ pW0 = W + (long)gn0 * K + ho;
    const float* __restrict__ pW1 = W + (long)gn1 * K + ho;
    const float* __restrict__ pA0 = A1 + rA0 + ho;
    const float* __restrict__ pA1 = A1 + rA1 + ho;

    f8 acc[2][2] = {{{}, {}}, {{}, {}}};

    // ---- phase 1: K in [0, K1), all from A1 (branch-free) ----
#pragma unroll 2
    for (int kc = 0; kc < K1; kc += 32) {
        __builtin_prefetch(pW0 + kc + 64, 0, 1);
        __builtin_prefetch(pW1 + kc + 64, 0, 1);
        wmma_chunk(acc, pA0 + kc, pA1 + kc, pW0 + kc, pW1 + kc);
    }
    // ---- phase 2: K in [K1, K), all from A2 (branch-free) ----
    if (A2) {
        const float* __restrict__ qA0 = A2 + rB0 + ho;
        const float* __restrict__ qA1 = A2 + rB1 + ho;
        const int K2f = K2 & ~31;
        int k2 = 0;
#pragma unroll 2
        for (; k2 < K2f; k2 += 32) {
            __builtin_prefetch(pW0 + K1 + k2 + 64, 0, 1);
            wmma_chunk(acc, qA0 + k2, qA1 + k2, pW0 + K1 + k2, pW1 + K1 + k2);
        }
        if (k2 < K2)   // K2 % 32 == 16 tail (uniform, at most once)
            wmma_chunk_half(acc, qA0 + k2, qA1 + k2, pW0 + K1 + k2, pW1 + K1 + k2);
    }

#pragma unroll
    for (int mi = 0; mi < 2; ++mi) {
#pragma unroll
        for (int ni = 0; ni < 2; ++ni) {
            const int gnc = tN * 32 + ni * 16 + ml;
#pragma unroll
            for (int v = 0; v < 8; ++v) {
                const int gm = tM * 32 + mi * 16 + v + 8 * half;
                float val = acc[mi][ni][v];
                if (bias) val += bias[gnc];
                if (relu) val = fmaxf(val, 0.f);
                if (seg) {
                    __hip_atomic_fetch_add(&segOut[(long)seg[gm] * N + gnc], val,
                                           __ATOMIC_RELAXED, __HIP_MEMORY_SCOPE_AGENT);
                } else {
                    OUT[(long)gm * ldOut + outColOff + gnc] = val;
                }
            }
        }
    }
}

// ---------------- small utility kernels ----------------
__global__ void fill_f32(float* p, int n, float v) {
    int i = blockIdx.x * blockDim.x + threadIdx.x;
    if (i < n) p[i] = v;
}
__global__ void fill_i32(int* p, int n, int v) {
    int i = blockIdx.x * blockDim.x + threadIdx.x;
    if (i < n) p[i] = v;
}
__global__ void count_seg(const int* __restrict__ idx, int n, float* __restrict__ cnt) {
    int i = blockIdx.x * blockDim.x + threadIdx.x;
    if (i < n)
        __hip_atomic_fetch_add(&cnt[idx[i]], 1.0f,
                               __ATOMIC_RELAXED, __HIP_MEMORY_SCOPE_AGENT);
}
__global__ void seg_mean_div(float* __restrict__ sum, const float* __restrict__ cnt,
                             int rows, int cols) {
    int i = blockIdx.x * blockDim.x + threadIdx.x;
    if (i < rows * cols) sum[i] /= fmaxf(cnt[i / cols], 1.0f);
}
__global__ void copy_cols(const float* __restrict__ src, float* __restrict__ dst,
                          int rows, int cols, int ldsrc, int lddst) {
    int i = blockIdx.x * blockDim.x + threadIdx.x;
    if (i < rows * cols) {
        int r = i / cols, c = i % cols;
        dst[(long)r * lddst + c] = src[(long)r * ldsrc + c];
    }
}
__global__ void col_max(const float* __restrict__ T, float* __restrict__ out,
                        int rows, int cols) {
    int c = blockIdx.x * blockDim.x + threadIdx.x;
    if (c < cols) {
        float m = -__builtin_inff();
        for (int r = 0; r < rows; ++r) m = fmaxf(m, T[(long)r * cols + c]);
        out[c] = m;
    }
}
__global__ void build_padsrc(const int* __restrict__ e0, const int* __restrict__ e1,
                             int E, int* __restrict__ deg, int* __restrict__ pad) {
    int i = blockIdx.x * blockDim.x + threadIdx.x;
    if (i < E) {
        int d = e1[i];
        int r = atomicAdd(&deg[d], 1);
        if (r < MAXD) pad[(long)d * MAXD + r] = e0[i];
    }
}

// ---------------- decode helpers ----------------
__device__ __forceinline__ float sigf(float x) { return 1.0f / (1.0f + __expf(-x)); }
__device__ __forceinline__ float wred_max(float v) {
#pragma unroll
    for (int o = 16; o > 0; o >>= 1) v = fmaxf(v, __shfl_xor(v, o));
    return v;
}
__device__ __forceinline__ float wred_sum(float v) {
#pragma unroll
    for (int o = 16; o > 0; o >>= 1) v += __shfl_xor(v, o);
    return v;
}
__device__ __forceinline__ float dot4(float4 a, float4 b) {
    return a.x * b.x + a.y * b.y + a.z * b.z + a.w * b.w;
}
// JAX threefry2x32 (5 blocks of 4 rounds)
__device__ __forceinline__ void tf2x32(unsigned k0, unsigned k1,
                                       unsigned& x0, unsigned& x1) {
    unsigned ks[3] = {k0, k1, k0 ^ k1 ^ 0x1BD11BDAu};
    const int rotA[4] = {13, 15, 26, 6};
    const int rotB[4] = {17, 29, 16, 24};
    x0 += ks[0]; x1 += ks[1];
#pragma unroll
    for (int b = 0; b < 5; ++b) {
        const int* rot = (b & 1) ? rotB : rotA;
#pragma unroll
        for (int j = 0; j < 4; ++j) {
            x0 += x1;
            x1 = (x1 << rot[j]) | (x1 >> (32 - rot[j]));
            x1 ^= x0;
        }
        x0 += ks[(b + 1) % 3];
        x1 += ks[(b + 2) % 3] + (unsigned)(b + 1);
    }
}

// =====================================================================
// Persistent decoder: full 1024-step scan in one 1024-thread workgroup.
// =====================================================================
__global__ __launch_bounds__(1024, 1) void decode_kernel(
    const float* __restrict__ opsE,    // [1024,1024]
    const float* __restrict__ slotsE,  // [256,1024]
    const float* __restrict__ kk,      // [256,1024]
    const float* __restrict__ gE,      // [1024]  (initial h)
    const float* __restrict__ c0,      // [1024]
    const float* __restrict__ Wih,     // [4096,1024]
    const float* __restrict__ Whh,     // [4096,1024]
    const float* __restrict__ bih,     // [4096]
    const float* __restrict__ bhh,     // [4096]
    const float* __restrict__ wq,      // [1024,2048]
    const int*   __restrict__ pad_src, // [1024,128]
    float* __restrict__ out)           // [2048]: placement(float), log_probs
{
    __shared__ float sh[EMBED], sc[EMBED], sinp[EMBED];
    __shared__ float sg[4 * EMBED];
    __shared__ float sal[N_OPS], sctx[EMBED], sq[EMBED];
    __shared__ float su[N_SLOTS], rv[N_SLOTS];
    __shared__ int   ri[N_SLOTS];
    __shared__ int   plc[N_OPS];
    __shared__ int   ssrc[MAXD];
    __shared__ unsigned sbits[N_SLOTS];
    __shared__ float sred[32];

    const int t = threadIdx.x;
    sh[t] = gE[t]; sc[t] = c0[t]; plc[t] = 0;
    __syncthreads();

    for (int i = 0; i < N_OPS; ++i) {
        if (t < MAXD) ssrc[t] = pad_src[(long)i * MAXD + t];
        __syncthreads();

        // ---- last = max over valid upstream slots; inp = ops[i] + last ----
        {
            float mx = -__builtin_inff();
            bool any = false;
            for (int d = 0; d < MAXD; ++d) {
                int s = ssrc[d];
                if (s >= 0) {
                    any = true;
                    mx = fmaxf(mx, slotsE[(long)plc[s] * EMBED + t]);
                }
            }
            sinp[t] = opsE[(long)i * EMBED + t] + (any ? mx : 0.f);
        }
        __syncthreads();

        // ---- g = inp@Wih^T + bih + h@Whh^T + bhh  (4 rows / thread) ----
#pragma unroll
        for (int j = 0; j < 4; ++j) {
            const int r = t + (j << 10);
            const float4* wi = (const float4*)(Wih + (long)r * EMBED);
            const float4* wh = (const float4*)(Whh + (long)r * EMBED);
            const float4* xi = (const float4*)sinp;
            const float4* xh = (const float4*)sh;
            float acc = bih[r] + bhh[r];
#pragma unroll 2
            for (int k = 0; k < EMBED / 4; ++k)
                acc += dot4(xi[k], wi[k]) + dot4(xh[k], wh[k]);
            sg[r] = acc;
        }
        __syncthreads();

        // ---- LSTM gates ----
        {
            float ig = sg[t], fg = sg[1024 + t], gg = sg[2048 + t], og = sg[3072 + t];
            float cc = sigf(fg) * sc[t] + sigf(ig) * tanhf(gg);
            sc[t] = cc;
            sh[t] = sigf(og) * tanhf(cc);
        }
        __syncthreads();

        // ---- attention scores + softmax over 1024 ops ----
        {
            const float4* row = (const float4*)(opsE + (long)t * EMBED);
            const float4* hv  = (const float4*)sh;
            float s = 0.f;
#pragma unroll 2
            for (int k = 0; k < EMBED / 4; ++k) s += dot4(row[k], hv[k]);
            sal[t] = s;
        }
        __syncthreads();
        {
            float v = sal[t];
            float wm = wred_max(v);
            if ((t & 31) == 0) sred[t >> 5] = wm;
            __syncthreads();
            if (t < 32) { float r = wred_max(sred[t]); if (t == 0) sred[0] = r; }
            __syncthreads();
            float mx = sred[0];
            __syncthreads();
            float e = __expf(v - mx);
            float wsum = wred_sum(e);
            if ((t & 31) == 0) sred[t >> 5] = wsum;
            __syncthreads();
            if (t < 32) { float r = wred_sum(sred[t]); if (t == 0) sred[0] = r; }
            __syncthreads();
            sal[t] = e / sred[0];
        }
        __syncthreads();

        // ---- ctx[c] = sum_n alpha[n] * opsE[n][c] (coalesced over c) ----
        {
            float acc = 0.f;
            for (int n = 0; n < N_OPS; ++n) acc += sal[n] * opsE[(long)n * EMBED + t];
            sctx[t] = acc;
        }
        __syncthreads();

        // ---- q = concat(h, ctx) @ wq^T ----
        {
            const float4* wr = (const float4*)(wq + (long)t * (2 * EMBED));
            const float4* hv = (const float4*)sh;
            const float4* cv = (const float4*)sctx;
            float acc = 0.f;
#pragma unroll 2
            for (int k = 0; k < EMBED / 4; ++k) acc += dot4(hv[k], wr[k]);
#pragma unroll 2
            for (int k = 0; k < EMBED / 4; ++k) acc += dot4(cv[k], wr[EMBED / 4 + k]);
            sq[t] = acc;
        }
        __syncthreads();

        // ---- slot logits u = 10*tanh((kk @ q)/32) ----
        if (t < N_SLOTS) {
            const float4* kr = (const float4*)(kk + (long)t * EMBED);
            const float4* qv = (const float4*)sq;
            float acc = 0.f;
#pragma unroll 2
            for (int k = 0; k < EMBED / 4; ++k) acc += dot4(kr[k], qv[k]);
            su[t] = 10.0f * tanhf(acc * (1.0f / 32.0f));
        }
        __syncthreads();

        // ---- JAX PRNG: fold_in(key(42), i) then 256 random bits ----
        if (t < 128) {
            unsigned d0 = 0u, d1 = (unsigned)i;
            tf2x32(0u, 42u, d0, d1);             // folded key = (d0,d1)
            unsigned b0 = (unsigned)t, b1 = (unsigned)(128 + t);
            tf2x32(d0, d1, b0, b1);
            sbits[t] = b0; sbits[128 + t] = b1;
        }
        __syncthreads();

        // ---- log-sum-exp of u (for log_prob) ----
        if (t < N_SLOTS) rv[t] = su[t];
        __syncthreads();
        for (int s = 128; s > 0; s >>= 1) {
            if (t < s) rv[t] = fmaxf(rv[t], rv[t + s]);
            __syncthreads();
        }
        float mu = rv[0];
        __syncthreads();
        if (t < N_SLOTS) rv[t] = __expf(su[t] - mu);
        __syncthreads();
        for (int s = 128; s > 0; s >>= 1) {
            if (t < s) rv[t] += rv[t + s];
            __syncthreads();
        }
        float lse = mu + __logf(rv[0]);
        __syncthreads();

        // ---- gumbel-argmax categorical sample (tie -> lowest index) ----
        if (t < N_SLOTS) {
            unsigned bits = sbits[t];
            float u01 = __uint_as_float((bits >> 9) | 0x3f800000u) - 1.0f;
            const float tiny = 1.17549435e-38f;
            float uu = u01 * (1.0f - tiny) + tiny;
            float gum = -__logf(-__logf(uu));
            rv[t] = su[t] + gum;
            ri[t] = t;
        }
        __syncthreads();
        for (int s = 128; s > 0; s >>= 1) {
            if (t < s) {
                float v2 = rv[t + s]; int i2 = ri[t + s];
                if (v2 > rv[t] || (v2 == rv[t] && i2 < ri[t])) { rv[t] = v2; ri[t] = i2; }
            }
            __syncthreads();
        }
        if (t == 0) {
            int a = ri[0];
            plc[i] = a;
            out[i] = (float)a;
            out[N_OPS + i] = su[a] - lse;
        }
        __syncthreads();
    }
}

// =====================================================================
// Host orchestration
// =====================================================================
extern "C" void kernel_launch(void* const* d_in, const int* in_sizes, int n_in,
                              void* d_out, int out_size, void* d_ws, size_t ws_size,
                              hipStream_t stream) {
    (void)in_sizes; (void)n_in; (void)out_size; (void)ws_size;

    // inputs in setup_inputs() insertion order (params flattened in insertion order)
    const float* op_feats   = (const float*)d_in[0];
    const int*   dsp        = (const int*)d_in[1];
    const float* slot_feats = (const float*)d_in[2];
    const int*   res        = (const int*)d_in[3];
    const float* eattr      = (const float*)d_in[4];
    const float* up1W  = (const float*)d_in[5];
    const float* up1b  = (const float*)d_in[6];
    const float* up1Wu = (const float*)d_in[7];
    const float* dn1W  = (const float*)d_in[8];
    const float* dn1b  = (const float*)d_in[9];
    const float* dn1Wu = (const float*)d_in[10];
    const float* up2W  = (const float*)d_in[11];
    const float* up2b  = (const float*)d_in[12];
    const float* up2Wu = (const float*)d_in[13];
    // d_in[14..16] = down2 (unused by reference — up_conv2 used twice)
    const float* r1W   = (const float*)d_in[17];
    const float* r1b   = (const float*)d_in[18];
    const float* r1Wu  = (const float*)d_in[19];
    const float* r2W   = (const float*)d_in[20];
    const float* r2b   = (const float*)d_in[21];
    const float* r2Wu  = (const float*)d_in[22];
    const float* Wih   = (const float*)d_in[23];
    const float* Whh   = (const float*)d_in[24];
    const float* bih   = (const float*)d_in[25];
    const float* bhh   = (const float*)d_in[26];
    const float* cell0 = (const float*)d_in[27];
    const float* aggW  = (const float*)d_in[28];
    const float* aggb  = (const float*)d_in[29];
    const float* wqW   = (const float*)d_in[30];
    const float* wkW   = (const float*)d_in[31];

    const int* e0 = dsp;
    const int* e1 = dsp + E_DSP;
    const int* r0 = res;
    const int* r1 = res + E_RES;

    // workspace carve (floats)
    float* ws = (float*)d_ws;
    float* xbuf   = ws;                         // [1024,1024]
    float* opsE   = xbuf   + N_OPS * EMBED;     // [1024,1024]
    float* aggbuf = opsE   + N_OPS * EMBED;     // [1024,512]
    float* Tbuf   = aggbuf + N_OPS * HALF_D;    // [1024,1024]
    float* s1     = Tbuf   + N_OPS * EMBED;     // [256,1024]
    float* slotsE = s1     + N_SLOTS * EMBED;   // [256,1024]
    float* raggr  = slotsE + N_SLOTS * EMBED;   // [256,1024]
    float* kkb    = raggr  + N_SLOTS * EMBED;   // [256,1024]
    float* gEb    = kkb    + N_SLOTS * EMBED;   // [1024]
    float* cnt    = gEb    + EMBED;             // [1024]
    int*   padv   = (int*)(cnt + N_OPS);        // [1024,128]
    int*   deg    = padv + N_OPS * MAXD;        // [1024]

    auto zf = [&](float* p, int n, float v) {
        fill_f32<<<(n + 255) / 256, 256, 0, stream>>>(p, n, v);
    };
    auto gemm = [&](const float* A1, int K1, const float* A2, int K2,
                    const int* gidx, const float* W, const float* bias,
                    int M, int N, float* OUT, int ld, int off,
                    const int* seg, float* segOut, int relu) {
        int tiles = (M / 32) * (N / 32);
        wmma_gemm<<<(tiles + 7) / 8, 256, 0, stream>>>(
            A1, K1, A2, K2, gidx, W, bias, M, N, OUT, ld, off, seg, segOut, relu);
    };

    // ---- DAG layer 1 (up) ----
    zf(cnt, N_OPS, 0.f);
    count_seg<<<E_DSP / 256, 256, 0, stream>>>(e1, E_DSP, cnt);
    zf(aggbuf, N_OPS * HALF_D, 0.f);
    gemm(op_feats, OP_DIM, nullptr, 0, e0, up1W, up1b, E_DSP, HALF_D,
         nullptr, 0, 0, e1, aggbuf, 1);
    seg_mean_div<<<(N_OPS * HALF_D + 255) / 256, 256, 0, stream>>>(aggbuf, cnt, N_OPS, HALF_D);
    gemm(op_feats, OP_DIM, aggbuf, HALF_D, nullptr, up1Wu, nullptr, N_OPS, HALF_D,
         xbuf, EMBED, 0, nullptr, nullptr, 1);

    // ---- DAG layer 1 (down) ----
    zf(cnt, N_OPS, 0.f);
    count_seg<<<E_DSP / 256, 256, 0, stream>>>(e0, E_DSP, cnt);
    zf(aggbuf, N_OPS * HALF_D, 0.f);
    gemm(op_feats, OP_DIM, nullptr, 0, e1, dn1W, dn1b, E_DSP, HALF_D,
         nullptr, 0, 0, e0, aggbuf, 1);
    seg_mean_div<<<(N_OPS * HALF_D + 255) / 256, 256, 0, stream>>>(aggbuf, cnt, N_OPS, HALF_D);
    gemm(op_feats, OP_DIM, aggbuf, HALF_D, nullptr, dn1Wu, nullptr, N_OPS, HALF_D,
         xbuf, EMBED, HALF_D, nullptr, nullptr, 1);

    // ---- DAG layer 2 (up2 used twice -> compute once, duplicate) ----
    zf(cnt, N_OPS, 0.f);
    count_seg<<<E_DSP / 256, 256, 0, stream>>>(e1, E_DSP, cnt);
    zf(aggbuf, N_OPS * HALF_D, 0.f);
    gemm(xbuf, EMBED, nullptr, 0, e0, up2W, up2b, E_DSP, HALF_D,
         nullptr, 0, 0, e1, aggbuf, 1);
    seg_mean_div<<<(N_OPS * HALF_D + 255) / 256, 256, 0, stream>>>(aggbuf, cnt, N_OPS, HALF_D);
    gemm(xbuf, EMBED, aggbuf, HALF_D, nullptr, up2Wu, nullptr, N_OPS, HALF_D,
         opsE, EMBED, 0, nullptr, nullptr, 1);
    copy_cols<<<(N_OPS * HALF_D + 255) / 256, 256, 0, stream>>>(
        opsE, opsE + HALF_D, N_OPS, HALF_D, EMBED, EMBED);

    // ---- graph embedding: max over relu(opsE @ aggW^T + aggb) ----
    gemm(opsE, EMBED, nullptr, 0, nullptr, aggW, aggb, N_OPS, EMBED,
         Tbuf, EMBED, 0, nullptr, nullptr, 1);
    col_max<<<(EMBED + 255) / 256, 256, 0, stream>>>(Tbuf, gEb, N_OPS, EMBED);

    // ---- res conv 1 ----
    zf(cnt, N_SLOTS, 0.f);
    count_seg<<<E_RES / 256, 256, 0, stream>>>(r1, E_RES, cnt);
    zf(raggr, N_SLOTS * EMBED, 0.f);
    gemm(slot_feats, SLOT_DIM, eattr, EDGE_DIM, r0, r1W, r1b, E_RES, EMBED,
         nullptr, 0, 0, r1, raggr, 1);
    seg_mean_div<<<(N_SLOTS * EMBED + 255) / 256, 256, 0, stream>>>(raggr, cnt, N_SLOTS, EMBED);
    gemm(slot_feats, SLOT_DIM, raggr, EMBED, nullptr, r1Wu, nullptr, N_SLOTS, EMBED,
         s1, EMBED, 0, nullptr, nullptr, 1);

    // ---- res conv 2 (cnt for r1 still valid) ----
    zf(raggr, N_SLOTS * EMBED, 0.f);
    gemm(s1, EMBED, eattr, EDGE_DIM, r0, r2W, r2b, E_RES, EMBED,
         nullptr, 0, 0, r1, raggr, 1);
    seg_mean_div<<<(N_SLOTS * EMBED + 255) / 256, 256, 0, stream>>>(raggr, cnt, N_SLOTS, EMBED);
    gemm(s1, EMBED, raggr, EMBED, nullptr, r2Wu, nullptr, N_SLOTS, EMBED,
         slotsE, EMBED, 0, nullptr, nullptr, 1);

    // ---- kk = slotsE @ wk^T ----
    gemm(slotsE, EMBED, nullptr, 0, nullptr, wkW, nullptr, N_SLOTS, EMBED,
         kkb, EMBED, 0, nullptr, nullptr, 0);

    // ---- padded upstream table ----
    fill_i32<<<(N_OPS * MAXD + 255) / 256, 256, 0, stream>>>(padv, N_OPS * MAXD, -1);
    fill_i32<<<(N_OPS + 255) / 256, 256, 0, stream>>>(deg, N_OPS, 0);
    build_padsrc<<<E_DSP / 256, 256, 0, stream>>>(e0, e1, E_DSP, deg, padv);

    // ---- persistent decoder ----
    decode_kernel<<<1, 1024, 0, stream>>>(opsE, slotsE, kkb, gEb, cell0,
                                          Wih, Whh, bih, bhh, wqW, padv,
                                          (float*)d_out);
}
